// TransformerBlock_25572235280880
// MI455X (gfx1250) — compile-verified, matrix operand order checked by reading, not run on hardware
//
#include <hip/hip_runtime.h>
#include <hip/hip_bf16.h>

// CDNA5 / gfx1250: all GEMMs via v_wmma_f32_16x16x32_bf16 (bf16 A/B, f32 acc),
// operands pre-swizzled into WMMA register layout (ISA 7.12.2), TDM staging of
// window tiles (ISA ch.8 Tensor Data Mover), wave32 shuffles for softmax.

typedef __attribute__((ext_vector_type(16))) __bf16 v16bf;
typedef __attribute__((ext_vector_type(8)))  float  v8f;
typedef __attribute__((ext_vector_type(4)))  unsigned v4u;
typedef __attribute__((ext_vector_type(8)))  int   v8i_;
typedef __attribute__((ext_vector_type(4)))  int   v4i_;

#define DEVINL __device__ __forceinline__

DEVINL __bf16 tobf(float f) { return (__bf16)f; }          // native v_cvt
DEVINL float frombf(__bf16 b) { return (float)b; }
DEVINL v8f zero8() { v8f z = {0.f,0.f,0.f,0.f,0.f,0.f,0.f,0.f}; return z; }

// (m, k) -> (lane, slot) for 16-bit A tile 16x32 (ISA 7.12.2); B tile uses the
// same k->(half,slot) map with lane = n + 16*half.
DEVINL void amap(int m, int kk, int& lane, int& e) {
  int half = (kk >> 3) & 1;
  lane = m + (half << 4);
  int v = ((kk >> 4) << 2) + ((kk & 7) >> 1);
  e = (v << 1) + (kk & 1);
}
DEVINL int kmap(int e, int half) {   // inverse: slot -> k
  int v = e >> 1, hi = e & 1;
  return ((v >> 2) << 4) + (half << 3) + ((v & 3) << 1) + hi;
}

// one contiguous 32B vector load of a pre-swizzled tile ([32][16] bf16)
DEVINL v16bf ldt(const __bf16* base) {
  int lane = threadIdx.x & 31;
  return *(const v16bf*)(base + lane * 16);
}
// pre-packed weight tile in global memory: [kt][nt][lane][16]
DEVINL v16bf ldpk(const __bf16* pk, int ntiles, int kt, int nt) {
  int lane = threadIdx.x & 31;
  return *(const v16bf*)(pk + (((size_t)kt * ntiles + nt) * 32 + lane) * 16);
}

DEVINL v8f wmma_bf(v16bf a, v16bf b, v8f c) {
  return __builtin_amdgcn_wmma_f32_16x16x32_bf16(false, a, false, b, (short)0, c,
                                                 false, false);
}
DEVINL float gelu_exact(float x) { return 0.5f * x * (1.f + erff(x * 0.70710678118f)); }

// C=64 NH=4 HEAD=16 WS=8 SS=4 N=64 H=W=Hp=Wp=256 nW=1024 E=3

// ------------------------------------------------------------- K0: pos table
__global__ void pos_kernel(float* __restrict__ pos) {
  int t = threadIdx.x;
  if (t >= 64) return;
  int ty = t >> 3, tx = t & 7;
  float sc = 2.f * 3.14159265358979f;
  float yv = (float)(ty + 1) / (8.f + 1e-6f) * sc;
  float xv = (float)(tx + 1) / (8.f + 1e-6f) * sc;
  for (int j = 0; j < 16; ++j) {
    float dim = powf(10000.f, (float)j / 16.f);
    pos[t * 64 + 2 * j]          = sinf(yv / dim);
    pos[t * 64 + 2 * j + 1]      = cosf(yv / dim);
    pos[t * 64 + 32 + 2 * j]     = sinf(xv / dim);
    pos[t * 64 + 32 + 2 * j + 1] = cosf(xv / dim);
  }
}

// ------------------------- K0b: repack f32 weight (KxN) -> bf16 WMMA-B tiles
__global__ void repack_kernel(const float* __restrict__ src, __bf16* __restrict__ dst,
                              int K, int N) {
  int idx = blockIdx.x * 256 + threadIdx.x;
  if (idx >= K * N) return;
  int k = idx / N, n = idx % N;
  int kt = k >> 5, kk = k & 31, nt = n >> 4, n16 = n & 15;
  int lane, e;
  amap(n16, kk, lane, e);                 // B: lane = n16 + 16*half(k)
  dst[(((size_t)kt * (N >> 4) + nt) * 32 + lane) * 16 + e] = tobf(src[idx]);
}

// ---------------------------------------------- K1: window + LN1 + pos + QKV
// grid 3*1024 blocks, 256 threads. qkv ws layout: [e][p][w][token][cc] bf16.
__global__ __launch_bounds__(256) void qkv_kernel(
    const float* __restrict__ x, const float* __restrict__ ln1w,
    const float* __restrict__ ln1b, const float* __restrict__ projp,
    const float* __restrict__ pos, __bf16* __restrict__ qkv) {
  __shared__ float sXfT[64][64];                       // [c][token]
  __shared__ __align__(32) __bf16 sA[4][2][32][16];    // A tiles (mt, kt)
  int blk = blockIdx.x;
  int e = blk >> 10, w = blk & 1023;
  int wy = w >> 5, wx = w & 31;
  int t = threadIdx.x;
  bool interior = (wy < 31) && (wx < 31);              // no roll wrap-around
  if (interior) {
    // Tensor Data Mover: 3D tile (x=8 contig, y=8 stride 256, z=64ch stride 64K)
    if (t < 32) {
      unsigned long long gaddr = (unsigned long long)(const void*)(
          x + ((size_t)(e * 64) * 256 + (wy * 8 + 4)) * 256 + (wx * 8 + 4));
      unsigned lds_off = (unsigned)(unsigned long long)(void*)&sXfT[0][0];
      v4u g0 = { 1u, lds_off, (unsigned)(gaddr & 0xFFFFFFFFu),
                 (unsigned)(((gaddr >> 32) & 0x01FFFFFFu) | 0x80000000u) };
      // data_size=4B; tensor dims large (never OOB); tile 8x8x64;
      // stride0=256, stride1=65536 elements
      v8i_ g1 = { 0x00020000, 0, 0x10, 0x00080010, 0x00400008, 256, 0, 1 };
      v4i_ g2 = { 1 << 20, 0, 65536, 0 };
      v4i_ g3 = { 0, 0, 0, 0 };
#if __clang_major__ >= 23
      v8i_ g4 = { 0, 0, 0, 0, 0, 0, 0, 0 };
      __builtin_amdgcn_tensor_load_to_lds(g0, g1, g2, g3, g4, 0);
#else
      __builtin_amdgcn_tensor_load_to_lds(g0, g1, g2, g3, 0);
#endif
      __builtin_amdgcn_s_wait_tensorcnt(0);
    }
  } else {
    for (int i = 0; i < 16; ++i) {
      int idx = t + i * 256;
      int c = idx >> 6, tok = idx & 63;
      int ty = tok >> 3, tx = tok & 7;
      int hs = (wy * 8 + ty + 4) & 255;                // roll(-SS) with wrap
      int wsv = (wx * 8 + tx + 4) & 255;
      sXfT[c][tok] = x[((size_t)(e * 64 + c) * 256 + hs) * 256 + wsv];
    }
  }
  __syncthreads();
  if (t < 64) {                                        // LN over 64 ch + pos
    float s = 0.f, ss = 0.f;
    for (int c = 0; c < 64; ++c) { float v = sXfT[c][t]; s += v; ss += v * v; }
    float m = s * (1.f / 64.f);
    float var = ss * (1.f / 64.f) - m * m;
    float inv = rsqrtf(var + 1e-5f);
    int mt = t >> 4, mrow = t & 15;
    for (int c = 0; c < 64; ++c) {
      float v = (sXfT[c][t] - m) * inv * ln1w[c] + ln1b[c] + pos[t * 64 + c];
      int lane, sl; amap(mrow, c & 31, lane, sl);
      sA[mt][c >> 5][lane][sl] = tobf(v);
    }
  }
  __syncthreads();
  int wv = t >> 5, lane = t & 31, half = lane >> 4, n16 = lane & 15;
  const __bf16* pj = (const __bf16*)projp;
  for (int i = 0; i < 6; ++i) {
    int tile = wv * 6 + i;                             // 4 M x 12 N tiles
    int mt = tile / 12, nt = tile % 12;
    v8f acc = zero8();
#pragma unroll
    for (int kk = 0; kk < 2; ++kk)
      acc = wmma_bf(ldt(&sA[mt][kk][0][0]), ldpk(pj, 12, kk, nt), acc);
#pragma unroll
    for (int r = 0; r < 8; ++r) {
      int row = mt * 16 + r + half * 8;                // token
      int col = nt * 16 + n16;                         // c3 in [0,192)
      int p = col >> 6, cc = col & 63;
      size_t o = ((size_t)((e * 3 + p) * 1024 + w) * 64 + row) * 64 + cc;
      qkv[o] = tobf(acc[r]);
    }
  }
}

// ----------------------------------------------------- K2: windowed attention
// grid 6*1024*4 blocks (combo, window, head), 64 threads (2 waves).
__global__ __launch_bounds__(64) void attn_kernel(const __bf16* __restrict__ qkv,
                                                  __bf16* __restrict__ aout) {
  __shared__ __align__(32) __bf16 sQA[4][32][16];      // Q A-tiles, K padded->32
  __shared__ __align__(32) __bf16 sKB[4][32][16];      // K^T B-tiles (nt)
  __shared__ __align__(32) __bf16 sVB[2][32][16];      // V B-tiles (kt)
  __shared__ __align__(32) __bf16 sAttA[4][2][32][16]; // softmaxed att A-tiles
  const int qi_[6] = {1, 1, 0, 0, 2, 2};
  const int ki_[6] = {0, 2, 1, 2, 1, 0};
  int blk = blockIdx.x;
  int g = blk >> 12;
  int rem = blk & 4095;
  int w = rem >> 2, h = rem & 3;
  int wy = w >> 5, wx = w & 31;
  int t = threadIdx.x;
  int eq = qi_[g], ek = ki_[g];
  const __bf16* qb = qkv + (size_t)((eq * 3 + 0) * 1024 + w) * 4096;
  const __bf16* kb = qkv + (size_t)((ek * 3 + 1) * 1024 + w) * 4096;
  const __bf16* vb = qkv + (size_t)((ek * 3 + 2) * 1024 + w) * 4096;
  // head split: channel cc = d*NH + h ; SCALE=0.25 folded into Q
  for (int i = 0; i < 32; ++i) {
    int idx = t + i * 64;                              // 0..2047
    int mt = idx >> 9, lane = (idx >> 4) & 31, e = idx & 15;
    int m = lane & 15, hf = lane >> 4, k = kmap(e, hf);
    float qv = (k < 16) ? frombf(qb[(mt * 16 + m) * 64 + k * 4 + h]) * 0.25f : 0.f;
    sQA[mt][lane][e] = tobf(qv);
    int nt = mt;                                       // same decomposition
    float kv = (k < 16) ? frombf(kb[(nt * 16 + m) * 64 + k * 4 + h]) : 0.f;
    sKB[nt][lane][e] = tobf(kv);                       // B: lane = n16 + 16*half
  }
  for (int i = 0; i < 16; ++i) {
    int idx = t + i * 64;                              // 0..1023
    int kt = idx >> 9, lane = (idx >> 4) & 31, e = idx & 15;
    int n = lane & 15, hf = lane >> 4, k = kt * 32 + kmap(e, hf);
    sVB[kt][lane][e] = vb[k * 64 + n * 4 + h];
  }
  __syncthreads();
  int wv = t >> 5, lane = t & 31, half = lane >> 4, n16 = lane & 15;
#pragma unroll
  for (int mi = 0; mi < 2; ++mi) {
    int mt = wv * 2 + mi;
    float vals[4][8];
    v16bf a = ldt(&sQA[mt][0][0]);
#pragma unroll
    for (int nt = 0; nt < 4; ++nt) {
      v8f acc = wmma_bf(a, ldt(&sKB[nt][0][0]), zero8());
#pragma unroll
      for (int r = 0; r < 8; ++r) {
        int qtok = mt * 16 + r + half * 8;
        int ktok = nt * 16 + n16;
        // Swin shift mask via region labels (thresholds Hp-WS=248, Hp-SS=252)
        int qhp = wy * 8 + (qtok >> 3), qwp = wx * 8 + (qtok & 7);
        int khp = wy * 8 + (ktok >> 3), kwp = wx * 8 + (ktok & 7);
        int lq = ((qhp < 248) ? 0 : ((qhp < 252) ? 1 : 2)) * 3 +
                 ((qwp < 248) ? 0 : ((qwp < 252) ? 1 : 2));
        int lk = ((khp < 248) ? 0 : ((khp < 252) ? 1 : 2)) * 3 +
                 ((kwp < 248) ? 0 : ((kwp < 252) ? 1 : 2));
        vals[nt][r] = acc[r] + ((lq != lk) ? -100.f : 0.f);
      }
    }
#pragma unroll
    for (int r = 0; r < 8; ++r) {                      // softmax per row
      float mx = fmaxf(fmaxf(vals[0][r], vals[1][r]), fmaxf(vals[2][r], vals[3][r]));
      for (int off = 1; off < 16; off <<= 1) mx = fmaxf(mx, __shfl_xor(mx, off, 32));
      float sum = 0.f;
#pragma unroll
      for (int nt = 0; nt < 4; ++nt) {
        float ev = __expf(vals[nt][r] - mx);
        vals[nt][r] = ev; sum += ev;
      }
      for (int off = 1; off < 16; off <<= 1) sum += __shfl_xor(sum, off, 32);
      float inv = 1.f / sum;
      int mrow = r + half * 8;
#pragma unroll
      for (int nt = 0; nt < 4; ++nt) {                 // write in A-tile layout
        int col = nt * 16 + n16;
        int lane2, e2; amap(mrow, col & 31, lane2, e2);
        sAttA[mt][col >> 5][lane2][e2] = tobf(vals[nt][r] * inv);
      }
    }
  }
  __syncthreads();
#pragma unroll
  for (int mi = 0; mi < 2; ++mi) {
    int mt = wv * 2 + mi;
    v8f acc = zero8();
#pragma unroll
    for (int kt = 0; kt < 2; ++kt)
      acc = wmma_bf(ldt(&sAttA[mt][kt][0][0]), ldt(&sVB[kt][0][0]), acc);
#pragma unroll
    for (int r = 0; r < 8; ++r) {
      int row = mt * 16 + r + half * 8;
      size_t o = ((size_t)(g * 1024 + w) * 64 + row) * 64 + (h * 16 + n16);
      aout[o] = tobf(acc[r]);                          // out channel = h*16 + d
    }
  }
}

// -------------------------- K3: exposure GEGLU + window reverse + residual
__global__ __launch_bounds__(256) void expmlp_kernel(
    const __bf16* __restrict__ aout, const __bf16* __restrict__ e11p,
    const float* __restrict__ e11b, const __bf16* __restrict__ e12p,
    const float* __restrict__ e12b, const __bf16* __restrict__ e2p,
    const float* __restrict__ e2b, const float* __restrict__ x,
    float* __restrict__ sc) {
  __shared__ __align__(32) __bf16 sA[4][4][32][16];    // inp A-tiles (K=128)
  __shared__ __align__(32) __bf16 sH[4][6][32][16];    // h A-tiles (K=192)
  int blk = blockIdx.x;
  int e = blk >> 10, w = blk & 1023;
  int wy = w >> 5, wx = w & 31;
  int t = threadIdx.x;
  const int gA_[3] = {2, 0, 4}, gB_[3] = {3, 1, 5};
  int gA = gA_[e], gB = gB_[e];
  for (int i = 0; i < 32; ++i) {
    int idx = t + i * 256;                             // 0..8191
    int mt = idx >> 11, kt = (idx >> 9) & 3, lane = (idx >> 4) & 31, sl = idx & 15;
    int m = lane & 15, hf = lane >> 4;
    int c2 = kt * 32 + kmap(sl, hf);
    int tok = mt * 16 + m;
    int g = (c2 < 64) ? gA : gB;
    sA[mt][kt][lane][sl] = aout[((size_t)(g * 1024 + w) * 64 + tok) * 64 + (c2 & 63)];
  }
  __syncthreads();
  int wv = t >> 5, lane = t & 31, half = lane >> 4, n16 = lane & 15;
  const __bf16* w11 = e11p + (size_t)e * 128 * 192;
  const __bf16* w12 = e12p + (size_t)e * 128 * 192;
  for (int i = 0; i < 6; ++i) {
    int tile = wv * 6 + i;                             // 4x12 tiles
    int mt = tile / 12, nt = tile % 12;
    v8f a1 = zero8(), a2 = zero8();
#pragma unroll
    for (int kk = 0; kk < 4; ++kk) {
      v16bf a = ldt(&sA[mt][kk][0][0]);
      a1 = wmma_bf(a, ldpk(w11, 12, kk, nt), a1);
      a2 = wmma_bf(a, ldpk(w12, 12, kk, nt), a2);
    }
#pragma unroll
    for (int r = 0; r < 8; ++r) {
      int mrow = r + half * 8;
      int col = nt * 16 + n16;
      float h1 = a1[r] + e11b[e * 192 + col];
      float h2 = a2[r] + e12b[e * 192 + col];
      int lane2, e2; amap(mrow, col & 31, lane2, e2);
      sH[mt][col >> 5][lane2][e2] = tobf(gelu_exact(h1) * h2);
    }
  }
  __syncthreads();
  const __bf16* w2 = e2p + (size_t)e * 192 * 64;
  for (int i = 0; i < 2; ++i) {
    int tile = wv * 2 + i;                             // 4x4 tiles
    int mt = tile >> 2, nt = tile & 3;
    v8f acc = zero8();
#pragma unroll
    for (int kk = 0; kk < 6; ++kk)
      acc = wmma_bf(ldt(&sH[mt][kk][0][0]), ldpk(w2, 4, kk, nt), acc);
#pragma unroll
    for (int r = 0; r < 8; ++r) {
      int tok = mt * 16 + r + half * 8;
      int cc = nt * 16 + n16;
      int ho = (wy * 8 + (tok >> 3) + 4) & 255;        // roll(+SS) back
      int wo = (wx * 8 + (tok & 7) + 4) & 255;
      size_t gi = ((size_t)(e * 64 + cc) * 256 + ho) * 256 + wo;
      sc[gi] = acc[r] + e2b[e * 64 + cc] + x[gi];      // + residual xo
    }
  }
}

// --------------------------------- K4: LN2 + big fused GEGLU (192->768->192)
__global__ __launch_bounds__(256) void mlp_kernel(
    const float* __restrict__ sc, const float* __restrict__ ln2w,
    const float* __restrict__ ln2b, const __bf16* __restrict__ m11p,
    const float* __restrict__ m11b, const __bf16* __restrict__ m12p,
    const float* __restrict__ m12b, const __bf16* __restrict__ m2p,
    const float* __restrict__ m2b, float* __restrict__ out) {
  __shared__ __align__(32) __bf16 sZ[4][6][32][16];    // LN'd rows, A (K=192)
  __shared__ __align__(32) __bf16 sHc[4][2][32][16];   // hidden chunk, A (K=64)
  int t = threadIdx.x;
  int p0 = blockIdx.x * 64;
  if (t < 64) {
    int p = p0 + t;
    float s = 0.f, ss = 0.f;
    for (int ch = 0; ch < 192; ++ch) {
      float v = sc[(size_t)ch * 65536 + p];
      s += v; ss += v * v;
    }
    float m = s * (1.f / 192.f);
    float var = ss * (1.f / 192.f) - m * m;
    float inv = rsqrtf(var + 1e-5f);
    int mt = t >> 4, mrow = t & 15;
    for (int ch = 0; ch < 192; ++ch) {
      float v = sc[(size_t)ch * 65536 + p];
      int lane2, e2; amap(mrow, ch & 31, lane2, e2);
      sZ[mt][ch >> 5][lane2][e2] = tobf((v - m) * inv * ln2w[ch] + ln2b[ch]);
    }
  }
  __syncthreads();
  int wv = t >> 5, lane = t & 31, half = lane >> 4, n16 = lane & 15;
  v8f acc[6];
#pragma unroll
  for (int i = 0; i < 6; ++i) acc[i] = zero8();
  for (int cn = 0; cn < 12; ++cn) {                    // 12 chunks of 64 cols
#pragma unroll
    for (int i = 0; i < 2; ++i) {                      // GEMM1: 4x4 tiles
      int tile = wv * 2 + i;
      int mt = tile >> 2, nt = tile & 3;
      v8f a1 = zero8(), a2 = zero8();
#pragma unroll
      for (int kk = 0; kk < 6; ++kk) {
        v16bf a = ldt(&sZ[mt][kk][0][0]);
        a1 = wmma_bf(a, ldpk(m11p, 48, kk, cn * 4 + nt), a1);
        a2 = wmma_bf(a, ldpk(m12p, 48, kk, cn * 4 + nt), a2);
      }
#pragma unroll
      for (int r = 0; r < 8; ++r) {
        int mrow = r + half * 8;
        int coln = nt * 16 + n16;
        float h1 = a1[r] + m11b[cn * 64 + coln];
        float h2 = a2[r] + m12b[cn * 64 + coln];
        int lane2, e2; amap(mrow, coln & 31, lane2, e2);
        sHc[mt][coln >> 5][lane2][e2] = tobf(gelu_exact(h1) * h2);
      }
    }
    __syncthreads();
#pragma unroll
    for (int i = 0; i < 6; ++i) {                      // GEMM2 acc: 4x12 tiles
      int tile = wv * 6 + i;
      int mt = tile / 12, nt = tile % 12;
#pragma unroll
      for (int kk = 0; kk < 2; ++kk)
        acc[i] = wmma_bf(ldt(&sHc[mt][kk][0][0]), ldpk(m2p, 12, cn * 2 + kk, nt),
                         acc[i]);
    }
    __syncthreads();
  }
#pragma unroll
  for (int i = 0; i < 6; ++i) {
    int tile = wv * 6 + i;
    int mt = tile / 12, nt = tile % 12;
#pragma unroll
    for (int r = 0; r < 8; ++r) {
      int pix = p0 + mt * 16 + r + half * 8;
      int ch = nt * 16 + n16;
      float z = acc[i][r] + m2b[ch];
      out[(size_t)ch * 65536 + pix] = sc[(size_t)ch * 65536 + pix] + z;
    }
  }
}

// ---------------------------------------------------------------- launcher
extern "C" void kernel_launch(void* const* d_in, const int* in_sizes, int n_in,
                              void* d_out, int out_size, void* d_ws, size_t ws_size,
                              hipStream_t stream) {
  (void)in_sizes; (void)n_in; (void)out_size;
  const float* x     = (const float*)d_in[0];
  const float* ln1w  = (const float*)d_in[1];
  const float* ln1b  = (const float*)d_in[2];
  const float* projw = (const float*)d_in[3];
  const float* e11w  = (const float*)d_in[4];
  const float* e11b  = (const float*)d_in[5];
  const float* e12w  = (const float*)d_in[6];
  const float* e12b  = (const float*)d_in[7];
  const float* e2w   = (const float*)d_in[8];
  const float* e2b   = (const float*)d_in[9];
  const float* ln2w  = (const float*)d_in[10];
  const float* ln2b  = (const float*)d_in[11];
  const float* m11w  = (const float*)d_in[12];
  const float* m11b  = (const float*)d_in[13];
  const float* m12w  = (const float*)d_in[14];
  const float* m12b  = (const float*)d_in[15];
  const float* m2w   = (const float*)d_in[16];
  const float* m2b   = (const float*)d_in[17];
  float* out = (float*)d_out;

  // workspace layout
  const size_t OFF_POS  = 0;                       // 16 KB f32 (64x64)
  const size_t OFF_QKV  = 64 * 1024;               // bf16, 75.5 MB
  const size_t OFF_AOUT = OFF_QKV + 75497472ull;   // bf16, 50.3 MB
  const size_t OFF_SC   = OFF_AOUT + 50331648ull;  // f32,  50.3 MB
  const size_t OFF_PJ   = OFF_SC + 50331648ull;    // packed bf16 weights:
  const size_t OFF_E11  = OFF_PJ + 24576;
  const size_t OFF_E12  = OFF_E11 + 147456;
  const size_t OFF_E2   = OFF_E12 + 147456;
  const size_t OFF_M11  = OFF_E2 + 73728;
  const size_t OFF_M12  = OFF_M11 + 294912;
  const size_t OFF_M2   = OFF_M12 + 294912;
  const size_t NEED     = OFF_M2 + 294912;
  if (ws_size < NEED) return;

  char* ws = (char*)d_ws;
  float*  pos  = (float*)(ws + OFF_POS);
  __bf16* qkv  = (__bf16*)(ws + OFF_QKV);
  __bf16* aout = (__bf16*)(ws + OFF_AOUT);
  float*  sc   = (float*)(ws + OFF_SC);
  __bf16* pjp  = (__bf16*)(ws + OFF_PJ);
  __bf16* e11p = (__bf16*)(ws + OFF_E11);
  __bf16* e12p = (__bf16*)(ws + OFF_E12);
  __bf16* e2p  = (__bf16*)(ws + OFF_E2);
  __bf16* m11p = (__bf16*)(ws + OFF_M11);
  __bf16* m12p = (__bf16*)(ws + OFF_M12);
  __bf16* m2p  = (__bf16*)(ws + OFF_M2);

  pos_kernel<<<1, 64, 0, stream>>>(pos);
  repack_kernel<<<48,  256, 0, stream>>>(projw, pjp, 64, 192);
  repack_kernel<<<288, 256, 0, stream>>>(e11w, e11p, 384, 192);  // 3x(128x192)
  repack_kernel<<<288, 256, 0, stream>>>(e12w, e12p, 384, 192);
  repack_kernel<<<144, 256, 0, stream>>>(e2w,  e2p,  576, 64);   // 3x(192x64)
  repack_kernel<<<576, 256, 0, stream>>>(m11w, m11p, 192, 768);
  repack_kernel<<<576, 256, 0, stream>>>(m12w, m12p, 192, 768);
  repack_kernel<<<576, 256, 0, stream>>>(m2w,  m2p,  768, 192);

  qkv_kernel<<<3072, 256, 0, stream>>>(x, ln1w, ln1b, (const float*)pjp, pos, qkv);
  attn_kernel<<<24576, 64, 0, stream>>>(qkv, aout);
  expmlp_kernel<<<3072, 256, 0, stream>>>(aout, e11p, e11b, e12p, e12b, e2p, e2b,
                                          x, sc);
  mlp_kernel<<<1024, 256, 0, stream>>>(sc, ln2w, ln2b, m11p, m11b, m12p, m12b,
                                       m2p, m2b, out);
}
